// BackboneR3Denoiser_87935160418352
// MI455X (gfx1250) — matrix operand order repaired and checked by reading, not compiled, not run on hardware
//
#include <hip/hip_runtime.h>

#define NN   1500
#define BBATCH 4
#define ESP  90000
#define ESL  12000
#define NLAY 8
#define CINQ 35
#define EDGE_BLOCKS 160   // persistent blocks for edge WMMA kernels (8 waves each)

typedef _Float16 half_t;
typedef __attribute__((ext_vector_type(16))) half_t v16h;
typedef __attribute__((ext_vector_type(8)))  float  v8f;

__device__ __forceinline__ float leakyf(float x){ return x > 0.f ? x : 0.2f * x; }
__device__ __forceinline__ unsigned encf(float f){
    unsigned u = __float_as_uint(f);
    return (u & 0x80000000u) ? ~u : (u | 0x80000000u);
}
__device__ __forceinline__ float decf(unsigned u){
    u = (u & 0x80000000u) ? (u & 0x7fffffffu) : ~u;
    return __uint_as_float(u);
}
// A-fragment K index for 16-bit 16x32 A tile (contiguous order, k = 16h + s, matching B)
__device__ __forceinline__ int kmapA(int s, int h){ return s + 16 * h; }

// ---------------- weight prep: f32 -> f16, Wa2 zero-padded 32x8 -> 32x16 ----------
__global__ void prep_weights_kernel(const float* __restrict__ Wa1, const float* __restrict__ Wa2,
                                    const float* __restrict__ We1, const float* __restrict__ We2,
                                    half_t* wa1h, half_t* wa2h, half_t* we1h, half_t* we2h){
    int t = blockIdx.x * blockDim.x + threadIdx.x;
    if (t < NLAY * 96 * 32) { wa1h[t] = (half_t)Wa1[t]; we1h[t] = (half_t)We1[t]; }
    if (t < NLAY * 32 * 32) we2h[t] = (half_t)We2[t];
    if (t < NLAY * 32 * 16) {
        int l = t / (32 * 16); int r = (t / 16) % 32; int c = t % 16;
        wa2h[t] = (half_t)(c < 8 ? Wa2[(l * 32 + r) * 8 + c] : 0.f);
    }
}

// ---------------- time MLP (B=4) ----------------
__global__ void time_mlp_kernel(const float* __restrict__ t, const float* __restrict__ kappa,
                                const float* __restrict__ W1, const float* __restrict__ b1,
                                const float* __restrict__ W2, const float* __restrict__ b2,
                                float* __restrict__ et){
    __shared__ float ft[64];
    __shared__ float h1[128];
    int b = blockIdx.x; int k = threadIdx.x; // 128 threads
    if (k < 32) {
        float tp = 6.283185307179586f * t[b] * kappa[k];
        ft[k] = cosf(tp); ft[32 + k] = sinf(tp);
    }
    __syncthreads();
    float s = b1[k];
    for (int c = 0; c < 64; ++c) s += ft[c] * W1[c * 128 + k];
    h1[k] = fmaxf(s, 0.f);
    __syncthreads();
    if (k < 64) {
        float s2 = b2[k];
        for (int c = 0; c < 128; ++c) s2 += h1[c] * W2[c * 64 + k];
        et[b * 64 + k] = fmaxf(s2, 0.f);
    }
}

// ---------------- edge features: rbf16 + sinusoidal(src-dst,16). R not needed. ----
__global__ void edge_feats_kernel(const int* __restrict__ ei, const float* __restrict__ nbb,
                                  float* __restrict__ ef, int E){
    int e = blockIdx.x * blockDim.x + threadIdx.x;
    if (e >= E) return;
    int s = ei[e], d = ei[E + e];
    float vx = nbb[(s * 4 + 1) * 3 + 0] - nbb[(d * 4 + 1) * 3 + 0];
    float vy = nbb[(s * 4 + 1) * 3 + 1] - nbb[(d * 4 + 1) * 3 + 1];
    float vz = nbb[(s * 4 + 1) * 3 + 2] - nbb[(d * 4 + 1) * 3 + 2];
    float dn = sqrtf(vx * vx + vy * vy + vz * vz);
    float* o = ef + (size_t)e * 32;
    #pragma unroll
    for (int j = 0; j < 16; ++j) {
        float u = (dn - j * 1.3333333333f) * 0.8f;
        o[j] = __expf(-u * u);
    }
    float diff = (float)(s - d);
    #pragma unroll
    for (int j = 0; j < 8; ++j) {
        float freq = __expf((2.f * j) * (-0.5756462732485114f)); // -ln(1e4)/16
        float ang = diff * freq;
        o[16 + j] = cosf(ang);
        o[24 + j] = sinf(ang);
    }
}

// ---------------- node init: sinusoid l0, embed MLP + LN, extraG (geometry ch) ----
__global__ void node_embed_kernel(const float* __restrict__ et, const int* __restrict__ batch,
                                  const float* __restrict__ W1, const float* __restrict__ b1,
                                  const float* __restrict__ W2, const float* __restrict__ b2,
                                  const float* __restrict__ g,  const float* __restrict__ bv,
                                  const float* __restrict__ nbb, const unsigned char* __restrict__ nmask,
                                  float* __restrict__ x, float* __restrict__ extraG){
    __shared__ float hin[96];
    __shared__ float h1[128];
    __shared__ float ush[32];
    int n = blockIdx.x; int k = threadIdx.x; // 128 threads
    if (k < 16) {
        float freq = __expf((2.f * k) * (-0.2878231366242557f)); // -ln(1e4)/32
        float ang = (float)n * freq;
        hin[k] = cosf(ang); hin[16 + k] = sinf(ang);
    }
    if (k < 64) hin[32 + k] = et[batch[n] * 64 + k];
    if (k < 12) {
        int i = k / 3, j = k % 3;
        float v;
        if (i == 0) v = (j == 2) ? (nmask[n] ? 1.f : 0.f) : 0.f;
        else {
            int atom = (j == 0) ? 0 : ((j == 1) ? 2 : 3);
            v = nbb[(n * 4 + atom) * 3 + (i - 1)];
        }
        extraG[n * 12 + k] = v;
    }
    __syncthreads();
    float s = b1[k];
    for (int c = 0; c < 96; ++c) s += hin[c] * W1[c * 128 + k];
    h1[k] = fmaxf(s, 0.f);
    __syncthreads();
    if (k < 32) {
        float s2 = b2[k];
        for (int c = 0; c < 128; ++c) s2 += h1[c] * W2[c * 32 + k];
        ush[k] = fmaxf(s2, 0.f);
    }
    __syncthreads();
    if (k < 32) {
        float mu = 0.f;
        for (int c = 0; c < 32; ++c) mu += ush[c];
        mu *= (1.f / 32.f);
        float var = 0.f;
        for (int c = 0; c < 32; ++c) { float dd = ush[c] - mu; var += dd * dd; }
        var *= (1.f / 32.f);
        float ln = (ush[k] - mu) * rsqrtf(var + 1e-5f) * g[k] + bv[k];
        x[(n * 4 + 0) * 32 + k] = hin[k] + ln;
        x[(n * 4 + 1) * 32 + k] = 0.f;
        x[(n * 4 + 2) * 32 + k] = 0.f;
        x[(n * 4 + 3) * 32 + k] = 0.f;
    }
}

// ---------------- per-layer clears ----------------
__global__ void clear_kernel(unsigned* __restrict__ amaxU, float* __restrict__ denom,
                             float* __restrict__ agg){
    int t = blockIdx.x * blockDim.x + threadIdx.x;
    if (t < NN * 8) { amaxU[t] = 0u; denom[t] = 0.f; }
    if (t < NN * 256) agg[t] = 0.f;
}

// ---------------- per-layer node precompute: Hsrc, Htgt, Vnode ----------------
__global__ void node_pre_kernel(const float* __restrict__ x, const float* __restrict__ extraG,
                                const float* __restrict__ Wsrc, const float* __restrict__ bsrc,
                                const float* __restrict__ Wtgt, const float* __restrict__ btgt,
                                const float* __restrict__ Wv,
                                float* __restrict__ Hs, float* __restrict__ Ht,
                                float* __restrict__ Vn){
    __shared__ float nf[4][4 * CINQ];
    int sub = threadIdx.x >> 6;          // 4 nodes / block
    int t = threadIdx.x & 63;
    int n = blockIdx.x * 4 + sub;
    for (int idx = t; idx < 4 * CINQ; idx += 64) {
        int i = idx / CINQ, c = idx % CINQ;
        nf[sub][idx] = (c < 32) ? x[(n * 4 + i) * 32 + c] : extraG[n * 12 + i * 3 + (c - 32)];
    }
    __syncthreads();
    const float* f = nf[sub];
    for (int i = 0; i < 4; ++i) {
        float s = 0.f;
        for (int c = 0; c < CINQ; ++c) s += f[i * CINQ + c] * Wv[c * 64 + t];
        Vn[(n * 4 + i) * 64 + t] = s;
    }
    if (t < 32) {
        float hs = bsrc[t], ht = btgt[t];
        for (int c = 0; c < CINQ; ++c) {
            float fv = f[c];
            hs += fv * Wsrc[c * 32 + t];
            ht += fv * Wtgt[c * 32 + t];
        }
        Hs[n * 32 + t] = hs;
        Ht[n * 32 + t] = ht;
    }
}

// ---------------- WMMA edge attention logits (persistent; weights hoisted) --------
// a = lrelu([Hs(src)|Ht(dst)|ef] @ Wa1 + ba1) @ Wa2 + ba2 ; atomic segment max
__global__ void __launch_bounds__(256) edge_logits_kernel(
        const int* __restrict__ src, const int* __restrict__ dst,
        const float* __restrict__ ef, const float* __restrict__ Hs, const float* __restrict__ Ht,
        const half_t* __restrict__ wa1h, const half_t* __restrict__ wa2h,
        const float* __restrict__ ba1, const float* __restrict__ ba2,
        float* __restrict__ abuf, unsigned* __restrict__ amaxU, int E){
    __shared__ half_t ain[8][16 * 96];
    __shared__ half_t hbuf[8][16 * 32];
    int wave = threadIdx.x >> 5, lane = threadIdx.x & 31;
    int row = lane & 15, half = lane >> 4;

    // hoist all weight B-fragments into registers (reused across tiles)
    v16h Bw[3][2], B2;
    #pragma unroll
    for (int kci = 0; kci < 3; ++kci) {
        #pragma unroll
        for (int s = 0; s < 16; ++s) {
            int K = kci * 32 + s + 16 * half;
            Bw[kci][0][s] = wa1h[K * 32 + row];
            Bw[kci][1][s] = wa1h[K * 32 + 16 + row];
        }
    }
    #pragma unroll
    for (int s = 0; s < 16; ++s) { int K = s + 16 * half; B2[s] = wa2h[K * 16 + row]; }
    float bias0 = ba1[row], bias1 = ba1[16 + row];

    int ntiles = E / 16;
    for (int tile = blockIdx.x * 8 + wave; tile < ntiles; tile += gridDim.x * 8) {
        int e0 = tile * 16 + row;
        int sI = src[e0], dI = dst[e0];
        #pragma unroll
        for (int cc = 0; cc < 48; ++cc) {
            int c = half * 48 + cc;
            float v = (c < 32) ? Hs[sI * 32 + c]
                    : (c < 64) ? Ht[dI * 32 + (c - 32)]
                               : ef[(size_t)e0 * 32 + (c - 64)];
            ain[wave][row * 96 + c] = (half_t)v;
        }
        v8f acc0 = {0.f,0.f,0.f,0.f,0.f,0.f,0.f,0.f};
        v8f acc1 = {0.f,0.f,0.f,0.f,0.f,0.f,0.f,0.f};
        #pragma unroll
        for (int kci = 0; kci < 3; ++kci) {
            v16h A;
            #pragma unroll
            for (int s = 0; s < 16; ++s) A[s] = ain[wave][row * 96 + kci * 32 + kmapA(s, half)];
            acc0 = __builtin_amdgcn_wmma_f32_16x16x32_f16(false, A, false, Bw[kci][0], (short)0, acc0, false, false);
            acc1 = __builtin_amdgcn_wmma_f32_16x16x32_f16(false, A, false, Bw[kci][1], (short)0, acc1, false, false);
        }
        #pragma unroll
        for (int r = 0; r < 8; ++r) {
            int M = r + 8 * half;
            hbuf[wave][M * 32 + row]      = (half_t)leakyf(acc0[r] + bias0);
            hbuf[wave][M * 32 + 16 + row] = (half_t)leakyf(acc1[r] + bias1);
        }
        v16h A2;
        #pragma unroll
        for (int s = 0; s < 16; ++s) A2[s] = hbuf[wave][row * 32 + kmapA(s, half)];
        v8f acc2 = {0.f,0.f,0.f,0.f,0.f,0.f,0.f,0.f};
        acc2 = __builtin_amdgcn_wmma_f32_16x16x32_f16(false, A2, false, B2, (short)0, acc2, false, false);
        if (row < 8) {
            int h = row;
            float b2v = ba2[h];
            #pragma unroll
            for (int r = 0; r < 8; ++r) {
                int M = r + 8 * half;
                int e = tile * 16 + M;
                float a = acc2[r] + b2v;
                abuf[(size_t)e * 8 + h] = a;
                atomicMax(&amaxU[dst[e] * 8 + h], encf(a));
            }
        }
    }
}

// ---------------- softmax numerator + denom ----------------
__global__ void edge_softmax_kernel(const int* __restrict__ dst, float* __restrict__ abuf,
                                    const unsigned* __restrict__ amaxU, float* __restrict__ denom,
                                    int E){
    int t = blockIdx.x * blockDim.x + threadIdx.x;
    if (t >= E * 8) return;
    int e = t >> 3, h = t & 7;
    int d = dst[e];
    float amax = decf(amaxU[d * 8 + h]);
    if (!(amax > -1e38f && amax < 1e38f)) amax = 0.f;  // isfinite guard (ref)
    float ea = __expf(abuf[t] - amax);
    abuf[t] = ea;
    atomicAdd(&denom[d * 8 + h], ea);
}

// ---------------- weighted scatter aggregate (rotation cancels: alpha * Vnode[src])
__global__ void edge_aggregate_kernel(const int* __restrict__ src, const int* __restrict__ dst,
                                      const float* __restrict__ abuf, const float* __restrict__ denom,
                                      const float* __restrict__ Vn, float* __restrict__ agg, int E){
    int t = blockIdx.x * blockDim.x + threadIdx.x;
    if (t >= E * 32) return;
    int e = t >> 5, r = t & 31, i = r >> 3, h = r & 7;
    int s = src[e], d = dst[e];
    float alpha = abuf[(size_t)e * 8 + h] / (denom[d * 8 + h] + 1e-9f);
    const float* vp = Vn + ((size_t)s * 4 + i) * 64 + h * 8;
    float* ap = agg + ((size_t)d * 4 + i) * 64 + h * 8;
    #pragma unroll
    for (int c = 0; c < 8; ++c) atomicAdd(ap + c, alpha * vp[c]);
}

// ---------------- node update: Wo + residual, FFN + gate ----------------
__global__ void node_update_kernel(const float* __restrict__ agg,
                                   const float* __restrict__ Wo,  const float* __restrict__ bo,
                                   const float* __restrict__ Wf1, const float* __restrict__ bf1,
                                   const float* __restrict__ Wf2, const float* __restrict__ bf2,
                                   const float* __restrict__ Wg,  const float* __restrict__ bg,
                                   float* __restrict__ x){
    __shared__ float l0sh[8][32];
    __shared__ float hsh[8][32];
    int sub = threadIdx.x >> 5, c = threadIdx.x & 31;
    int n = blockIdx.x * 8 + sub;
    bool ok = n < NN;
    float xn[4] = {0.f, 0.f, 0.f, 0.f};
    if (ok) {
        for (int i = 0; i < 4; ++i) {
            float s = (i == 0) ? bo[c] : 0.f;
            const float* ag = agg + ((size_t)n * 4 + i) * 64;
            for (int v = 0; v < 64; ++v) s += ag[v] * Wo[v * 32 + c];
            xn[i] = x[(n * 4 + i) * 32 + c] + s;
        }
    }
    l0sh[sub][c] = xn[0];
    __syncthreads();
    float hc = bf1[c], gc = bg[c];
    for (int k = 0; k < 32; ++k) {
        float l = l0sh[sub][k];
        hc += l * Wf1[k * 32 + c];
        gc += l * Wg[k * 32 + c];
    }
    hc = fmaxf(hc, 0.f);
    hsh[sub][c] = hc;
    float gate = 1.f / (1.f + __expf(-gc));
    __syncthreads();
    float fc = bf2[c];
    for (int k = 0; k < 32; ++k) fc += hsh[sub][k] * Wf2[k * 32 + c];
    if (ok) {
        x[(n * 4 + 0) * 32 + c] = xn[0] + fc;
        float sc = 1.f + gate;
        x[(n * 4 + 1) * 32 + c] = xn[1] * sc;
        x[(n * 4 + 2) * 32 + c] = xn[2] * sc;
        x[(n * 4 + 3) * 32 + c] = xn[3] * sc;
    }
}

// ---------------- WMMA edge MLP (persistent; weights hoisted) ---------------------
// en = ef + relu([x0(dst)|x0(src)|ef] @ We1 + be1) @ We2 + be2
__global__ void __launch_bounds__(256) edge_mlp_kernel(
        const int* __restrict__ src, const int* __restrict__ dst,
        float* __restrict__ ef, const float* __restrict__ x,
        const half_t* __restrict__ we1h, const half_t* __restrict__ we2h,
        const float* __restrict__ be1, const float* __restrict__ be2, int E){
    __shared__ half_t ain[8][16 * 96];
    __shared__ half_t hbuf[8][16 * 32];
    int wave = threadIdx.x >> 5, lane = threadIdx.x & 31;
    int row = lane & 15, half = lane >> 4;

    v16h Bw[3][2], B2a, B2b;
    #pragma unroll
    for (int kci = 0; kci < 3; ++kci) {
        #pragma unroll
        for (int s = 0; s < 16; ++s) {
            int K = kci * 32 + s + 16 * half;
            Bw[kci][0][s] = we1h[K * 32 + row];
            Bw[kci][1][s] = we1h[K * 32 + 16 + row];
        }
    }
    #pragma unroll
    for (int s = 0; s < 16; ++s) {
        int K = s + 16 * half;
        B2a[s] = we2h[K * 32 + row];
        B2b[s] = we2h[K * 32 + 16 + row];
    }
    float bias0 = be1[row], bias1 = be1[16 + row];
    float ob0 = be2[row], ob1 = be2[16 + row];

    int ntiles = E / 16;
    for (int tile = blockIdx.x * 8 + wave; tile < ntiles; tile += gridDim.x * 8) {
        int e0 = tile * 16 + row;
        int sI = src[e0], dI = dst[e0];
        #pragma unroll
        for (int cc = 0; cc < 48; ++cc) {
            int c = half * 48 + cc;
            float v = (c < 32) ? x[(dI * 4) * 32 + c]
                    : (c < 64) ? x[(sI * 4) * 32 + (c - 32)]
                               : ef[(size_t)e0 * 32 + (c - 64)];
            ain[wave][row * 96 + c] = (half_t)v;
        }
        v8f acc0 = {0.f,0.f,0.f,0.f,0.f,0.f,0.f,0.f};
        v8f acc1 = {0.f,0.f,0.f,0.f,0.f,0.f,0.f,0.f};
        #pragma unroll
        for (int kci = 0; kci < 3; ++kci) {
            v16h A;
            #pragma unroll
            for (int s = 0; s < 16; ++s) A[s] = ain[wave][row * 96 + kci * 32 + kmapA(s, half)];
            acc0 = __builtin_amdgcn_wmma_f32_16x16x32_f16(false, A, false, Bw[kci][0], (short)0, acc0, false, false);
            acc1 = __builtin_amdgcn_wmma_f32_16x16x32_f16(false, A, false, Bw[kci][1], (short)0, acc1, false, false);
        }
        #pragma unroll
        for (int r = 0; r < 8; ++r) {
            int M = r + 8 * half;
            hbuf[wave][M * 32 + row]      = (half_t)fmaxf(acc0[r] + bias0, 0.f);
            hbuf[wave][M * 32 + 16 + row] = (half_t)fmaxf(acc1[r] + bias1, 0.f);
        }
        v16h A2;
        #pragma unroll
        for (int s = 0; s < 16; ++s) A2[s] = hbuf[wave][row * 32 + kmapA(s, half)];
        v8f o0 = {0.f,0.f,0.f,0.f,0.f,0.f,0.f,0.f};
        v8f o1 = {0.f,0.f,0.f,0.f,0.f,0.f,0.f,0.f};
        o0 = __builtin_amdgcn_wmma_f32_16x16x32_f16(false, A2, false, B2a, (short)0, o0, false, false);
        o1 = __builtin_amdgcn_wmma_f32_16x16x32_f16(false, A2, false, B2b, (short)0, o1, false, false);
        #pragma unroll
        for (int r = 0; r < 8; ++r) {
            int M = r + 8 * half;
            size_t e = (size_t)(tile * 16 + M);
            ef[e * 32 + row]      += o0[r] + ob0;
            ef[e * 32 + 16 + row] += o1[r] + ob1;
        }
    }
}

// ---------------- output: center cancels -> den = noised_bb + mask * updates ------
__global__ void output_kernel(const float* __restrict__ x, const float* __restrict__ nbb,
                              const unsigned char* __restrict__ nmask,
                              const float* __restrict__ W_xca, const float* __restrict__ W_bb,
                              float* __restrict__ den){
    int n = blockIdx.x * blockDim.x + threadIdx.x;
    if (n >= NN) return;
    bool m = nmask[n] != 0;
    for (int k = 0; k < 3; ++k) {
        float u = 0.f;
        if (m) for (int c = 0; c < 32; ++c) u += x[(n * 4 + 1 + k) * 32 + c] * W_xca[c];
        den[(n * 4 + 1) * 3 + k] = nbb[(n * 4 + 1) * 3 + k] + u;
    }
    for (int r = 0; r < 3; ++r) {
        int atom = (r == 0) ? 0 : ((r == 1) ? 2 : 3);
        for (int k = 0; k < 3; ++k) {
            float u = 0.f;
            if (m) for (int c = 0; c < 32; ++c) u += x[(n * 4 + 1 + k) * 32 + c] * W_bb[c * 3 + r];
            den[(n * 4 + atom) * 3 + k] = nbb[(n * 4 + atom) * 3 + k] + u;
        }
    }
}

extern "C" void kernel_launch(void* const* d_in, const int* in_sizes, int n_in,
                              void* d_out, int out_size, void* d_ws, size_t ws_size,
                              hipStream_t stream) {
    (void)in_sizes; (void)n_in; (void)out_size; (void)ws_size;
    const float* nbb   = (const float*)d_in[0];
    const float* t     = (const float*)d_in[1];
    const int*   batch = (const int*)d_in[2];
    const int*   ei    = (const int*)d_in[3];
    const int*   sei   = (const int*)d_in[4];
    const unsigned char* nmask = (const unsigned char*)d_in[6]; // x_mask (d_in[5]) is dead
    const float* W_bb  = (const float*)d_in[7];
    const float* W_xca = (const float*)d_in[8];
    const float* emW1 = (const float*)d_in[9],  *emW2 = (const float*)d_in[10];
    const float* emb  = (const float*)d_in[11], *emb1 = (const float*)d_in[12];
    const float* emb2 = (const float*)d_in[13], *emg  = (const float*)d_in[14];
    const float* kappa = (const float*)d_in[15];
    const float* Wa1 = (const float*)d_in[16], *Wa2 = (const float*)d_in[17];
    const float* We1 = (const float*)d_in[18], *We2 = (const float*)d_in[19];
    const float* Wf1 = (const float*)d_in[20], *Wf2 = (const float*)d_in[21];
    const float* Wg  = (const float*)d_in[22], *Wo  = (const float*)d_in[23];
    const float* Wsrc = (const float*)d_in[24], *Wtgt = (const float*)d_in[25];
    const float* Wv  = (const float*)d_in[26];
    const float* ba1 = (const float*)d_in[27], *ba2 = (const float*)d_in[28];
    const float* be1 = (const float*)d_in[29], *be2 = (const float*)d_in[30];
    const float* bf1 = (const float*)d_in[31], *bf2 = (const float*)d_in[32];
    const float* bg  = (const float*)d_in[33], *bo  = (const float*)d_in[34];
    const float* bsrc = (const float*)d_in[35], *btgt = (const float*)d_in[36];
    const float* tmW1 = (const float*)d_in[37], *tmW2 = (const float*)d_in[38];
    const float* tmb1 = (const float*)d_in[39], *tmb2 = (const float*)d_in[40];

    char* wsp = (char*)d_ws;
    size_t off = 0;
    auto alloc = [&](size_t bytes) { void* p = wsp + off; off += (bytes + 255) & ~(size_t)255; return p; };
    float*  x      = (float*)alloc((size_t)NN * 4 * 32 * 4);
    float*  extraG = (float*)alloc((size_t)NN * 12 * 4);
    float*  Hs     = (float*)alloc((size_t)NN * 32 * 4);
    float*  Ht     = (float*)alloc((size_t)NN * 32 * 4);
    float*  Vn     = (float*)alloc((size_t)NN * 256 * 4);
    float*  efsp   = (float*)alloc((size_t)ESP * 32 * 4);
    float*  efsl   = (float*)alloc((size_t)ESL * 32 * 4);
    float*  abuf   = (float*)alloc((size_t)ESP * 8 * 4);
    unsigned* amaxU = (unsigned*)alloc((size_t)NN * 8 * 4);
    float*  denom  = (float*)alloc((size_t)NN * 8 * 4);
    float*  agg    = (float*)alloc((size_t)NN * 256 * 4);
    float*  et     = (float*)alloc((size_t)BBATCH * 64 * 4);
    half_t* wa1h   = (half_t*)alloc((size_t)NLAY * 96 * 32 * 2);
    half_t* wa2h   = (half_t*)alloc((size_t)NLAY * 32 * 16 * 2);
    half_t* we1h   = (half_t*)alloc((size_t)NLAY * 96 * 32 * 2);
    half_t* we2h   = (half_t*)alloc((size_t)NLAY * 32 * 32 * 2);

    prep_weights_kernel<<<(NLAY * 96 * 32 + 255) / 256, 256, 0, stream>>>(Wa1, Wa2, We1, We2,
                                                                          wa1h, wa2h, we1h, we2h);
    time_mlp_kernel<<<BBATCH, 128, 0, stream>>>(t, kappa, tmW1, tmb1, tmW2, tmb2, et);
    edge_feats_kernel<<<(ESP + 255) / 256, 256, 0, stream>>>(ei, nbb, efsp, ESP);
    edge_feats_kernel<<<(ESL + 255) / 256, 256, 0, stream>>>(sei, nbb, efsl, ESL);
    node_embed_kernel<<<NN, 128, 0, stream>>>(et, batch, emW1, emb1, emW2, emb2, emg, emb,
                                              nbb, nmask, x, extraG);

    for (int l = 0; l < NLAY; ++l) {
        const int* es; const int* ed; float* ef; int E;
        if ((l & 1) == 0) { es = ei;  ed = ei + ESP;  ef = efsp; E = ESP; }
        else              { es = sei; ed = sei + ESL; ef = efsl; E = ESL; }
        clear_kernel<<<(NN * 256 + 255) / 256, 256, 0, stream>>>(amaxU, denom, agg);
        node_pre_kernel<<<NN / 4, 256, 0, stream>>>(x, extraG,
                Wsrc + l * 35 * 32, bsrc + l * 32, Wtgt + l * 35 * 32, btgt + l * 32,
                Wv + l * 35 * 64, Hs, Ht, Vn);
        int tiles = E / 16;
        int lb = (tiles + 7) / 8;
        if (lb > EDGE_BLOCKS) lb = EDGE_BLOCKS;
        edge_logits_kernel<<<lb, 256, 0, stream>>>(es, ed, ef, Hs, Ht,
                wa1h + l * 96 * 32, wa2h + l * 32 * 16, ba1 + l * 32, ba2 + l * 8,
                abuf, amaxU, E);
        edge_softmax_kernel<<<(E * 8 + 255) / 256, 256, 0, stream>>>(ed, abuf, amaxU, denom, E);
        edge_aggregate_kernel<<<(E * 32 + 255) / 256, 256, 0, stream>>>(es, ed, abuf, denom, Vn, agg, E);
        node_update_kernel<<<(NN + 7) / 8, 256, 0, stream>>>(agg,
                Wo + l * 64 * 32, bo + l * 32, Wf1 + l * 1024, bf1 + l * 32,
                Wf2 + l * 1024, bf2 + l * 32, Wg + l * 1024, bg + l * 32, x);
        edge_mlp_kernel<<<lb, 256, 0, stream>>>(es, ed, ef, x,
                we1h + l * 96 * 32, we2h + l * 32 * 32, be1 + l * 32, be2 + l * 32, E);
    }
    output_kernel<<<(NN + 255) / 256, 256, 0, stream>>>(x, nbb, nmask, W_xca, W_bb, (float*)d_out);
}
